// VFNet_35897336660684
// MI455X (gfx1250) — matrix operand order repaired and checked by reading, not compiled, not run on hardware
//
#include <hip/hip_runtime.h>
#include <hip/hip_bf16.h>

typedef __attribute__((ext_vector_type(16))) _Float16 v16h;
typedef __attribute__((ext_vector_type(8)))  float    v8f;

#define TPAD 1088      // 1025 tokens padded to 17*64 -> every GEMM exactly tiled
#define NTOK 1025
#define DMODEL 256
#define BATCH 64
#define NCLS 1000

#define BM 64
#define BN 128
#define BK 32
#define ASTR 40          // halfs per A-tile row (padded)
#define NKP  (BK/2)      // 16 k-pairs in B tile
#define BSTRP (2*BN + 8) // 264 halfs per k-pair row (pair-interleaved)

#define FLAG_LRELU  1
#define FLAG_OUTF16 2
#define FLAG_OUT_PI 4    // f16 pair-interleaved output (for B-operand buffers)

// ------------- async global->LDS copy (CDNA5, tracked by ASYNCcnt) ------------------
// Generic shared pointers: low 32 bits == LDS byte offset (ISA aperture rule).

__device__ __forceinline__ void async_copy_b128(void* lds, const void* gaddr) {
    unsigned l = (unsigned)(size_t)lds;
    unsigned long long g = (unsigned long long)(size_t)gaddr;
    asm volatile("global_load_async_to_lds_b128 %0, %1, off"
                 :: "v"(l), "v"(g) : "memory");
}
__device__ __forceinline__ void wait_async0() {
    asm volatile("s_wait_asynccnt 0x0" ::: "memory");
}

// ------------- fragment loads (CDNA5 wave32 layouts, CDNA5 ISA) --------------------

union FragU { v16h v; unsigned int u[8]; };

// A 16x32 f16, contiguous k chosen here (ISA table order differs; B uses the same): lanes0-15 rows 0-15 K{0..15}; lanes16-31 K{16..31}
__device__ __forceinline__ v16h load_a_frag(const _Float16* As, int rowBase, int lane) {
    int r = lane & 15, hi = lane >> 4;
    FragU f;
#pragma unroll
    for (int i = 0; i < 8; ++i) {
        int k = 2 * i + hi * 16;
        f.u[i] = *(const unsigned int*)&As[(rowBase + r) * ASTR + k];
    }
    return f.v;
}

// B 32x16 f16 from pair-interleaved LDS: u[v] = {B[2kp][n], B[2kp+1][n]}, kp = hi*8+v
__device__ __forceinline__ v16h load_b_frag(const _Float16* Bs, int colBase, int lane) {
    int n = lane & 15, hi = lane >> 4;
    FragU f;
#pragma unroll
    for (int v = 0; v < 8; ++v)
        f.u[v] = *(const unsigned int*)&Bs[(hi * 8 + v) * BSTRP + 2 * (colBase + n)];
    return f.v;
}

#define WMMA(a, b, c) __builtin_amdgcn_wmma_f32_16x16x32_f16(false, a, false, b, (short)0, c, false, false)

// ------------- generic guard-free WMMA GEMM (f16 in, f32 acc) -----------------------
// B is pair-interleaved in global memory: element (k,n) at (k/2)*2N + 2n + (k&1).
// Out = alpha*A@B (+bias)(+extra)(+res)(lrelu). M%64==0, N%128==0, K%32==0.

__global__ __launch_bounds__(256) void gemm_h_kernel(
    const _Float16* __restrict__ A, int lda, long sA,
    const _Float16* __restrict__ Bm, long sB,
    void* __restrict__ OutV, int ldo, long sO,
    const float* __restrict__ res, long sR,
    const float* __restrict__ extra, long sE,
    const float* __restrict__ bias,
    int N, int K, float alpha, int flags)
{
    __shared__ __align__(16) _Float16 As[2][BM * ASTR];
    __shared__ __align__(16) _Float16 Bs[2][NKP * BSTRP];

    int tid = threadIdx.x, lane = tid & 31, wave = tid >> 5;
    int wm = wave >> 2, wn = wave & 3;                 // 2 x 4 waves -> 32x32 each
    int z = blockIdx.z;
    int rowBase = blockIdx.y * BM, colBase = blockIdx.x * BN;

    const _Float16* Ap = A + (size_t)z * sA + (size_t)rowBase * lda;
    const _Float16* Bp = Bm + (size_t)z * sB;

    int ar = tid >> 2, ac = (tid & 3) * 8;             // A stage: 64 rows x 4 oct-cols
    int bkp = tid >> 4, bn = (tid & 15) * 8;           // B stage: 16 kpairs x 16 oct-cols
    long twoN = 2L * N;

    v8f acc[2][2] = {};
    const int nk = K / BK;

    auto stage = [&](int kt, int b) {
        // A: 16B async copy per thread
        async_copy_b128(&As[b][ar * ASTR + ac], &Ap[(size_t)ar * lda + kt * BK + ac]);
        // B (PI layout): 32 contiguous bytes per thread -> two async b128
        const _Float16* g = &Bp[(size_t)(kt * NKP + bkp) * twoN + 2 * (colBase + bn)];
        _Float16* l = &Bs[b][bkp * BSTRP + 2 * bn];
        async_copy_b128(l, g);
        async_copy_b128(l + 8, g + 8);
    };

    stage(0, 0);
    int buf = 0;
    for (int kt = 0; kt < nk; ++kt) {
        wait_async0();
        __syncthreads();
        if (kt + 1 < nk) stage(kt + 1, buf ^ 1);
        v16h a0 = load_a_frag(As[buf], wm * 32, lane);
        v16h a1 = load_a_frag(As[buf], wm * 32 + 16, lane);
        v16h b0 = load_b_frag(Bs[buf], wn * 32, lane);
        v16h b1 = load_b_frag(Bs[buf], wn * 32 + 16, lane);
        acc[0][0] = WMMA(a0, b0, acc[0][0]);
        acc[0][1] = WMMA(a0, b1, acc[0][1]);
        acc[1][0] = WMMA(a1, b0, acc[1][0]);
        acc[1][1] = WMMA(a1, b1, acc[1][1]);
        buf ^= 1;
    }

    // epilogue: C layout VGPR j: lanes0-15 M=j, lanes16-31 M=8+j; N=lane&15
    int hi = lane >> 4, nn = lane & 15;
    size_t zo = (size_t)z * sO;
#pragma unroll
    for (int mi = 0; mi < 2; ++mi) {
#pragma unroll
        for (int ni = 0; ni < 2; ++ni) {
            v8f a = acc[mi][ni];
            int col = colBase + wn * 32 + ni * 16 + nn;
            float bv = bias ? bias[col] : 0.f;
#pragma unroll
            for (int j = 0; j < 8; ++j) {
                int row = rowBase + wm * 32 + mi * 16 + hi * 8 + j;
                size_t o = (size_t)row * ldo + col;
                float v = alpha * a[j] + bv;
                if (extra) v += extra[(size_t)z * sE + o];
                if (res)   v += res[(size_t)z * sR + o];
                if (flags & FLAG_LRELU) v = v > 0.f ? v : 0.01f * v;
                if (flags & FLAG_OUT_PI) {
                    size_t opi = ((size_t)(row >> 1) * ldo + col) * 2 + (row & 1);
                    ((_Float16*)OutV)[zo + opi] = (_Float16)v;
                } else if (flags & FLAG_OUTF16) {
                    ((_Float16*)OutV)[zo + o] = (_Float16)v;
                } else {
                    ((float*)OutV)[zo + o] = v;
                }
            }
        }
    }
}

// ------------- patch-embed gather GEMM: tok = im2row(x) @ wT + cb + pos -------------
// M=B*1024, K=768, N=256 (grid 2 x 1024). xh f16 row-major; wT f16 PI.

__global__ __launch_bounds__(256) void patch_embed_kernel(
    const _Float16* __restrict__ xh, const _Float16* __restrict__ wT,
    const float* __restrict__ cb, const float* __restrict__ pe,
    float* __restrict__ t)
{
    __shared__ __align__(16) _Float16 As[2][BM * ASTR];
    __shared__ __align__(16) _Float16 Bs[2][NKP * BSTRP];

    int tid = threadIdx.x, lane = tid & 31, wave = tid >> 5;
    int wm = wave >> 2, wn = wave & 3;
    int rowBase = blockIdx.y * BM, colBase = blockIdx.x * BN;

    int ar = tid >> 2, ac = (tid & 3) * 8;
    int bkp = tid >> 4, bn = (tid & 15) * 8;

    int gr = rowBase + ar;
    int b  = gr >> 10, p = gr & 1023;
    int gh = p >> 5, gw = p & 31;

    v8f acc[2][2] = {};
    const int nk = 768 / BK;

    auto stage = [&](int kt, int bb) {
        int k = kt * BK + ac;                 // multiple of 8 -> 8 contiguous q
        int ch = k >> 8, rem = k & 255;
        int pp = rem >> 4, qq = rem & 15;
        size_t addr = ((size_t)(b * 3 + ch) * 512 + gh * 16 + pp) * 512 + gw * 16 + qq;
        async_copy_b128(&As[bb][ar * ASTR + ac], &xh[addr]);
        const _Float16* g = &wT[(size_t)(kt * NKP + bkp) * (2 * DMODEL) + 2 * (colBase + bn)];
        _Float16* l = &Bs[bb][bkp * BSTRP + 2 * bn];
        async_copy_b128(l, g);
        async_copy_b128(l + 8, g + 8);
    };

    stage(0, 0);
    int buf = 0;
    for (int kt = 0; kt < nk; ++kt) {
        wait_async0();
        __syncthreads();
        if (kt + 1 < nk) stage(kt + 1, buf ^ 1);
        v16h a0 = load_a_frag(As[buf], wm * 32, lane);
        v16h a1 = load_a_frag(As[buf], wm * 32 + 16, lane);
        v16h b0 = load_b_frag(Bs[buf], wn * 32, lane);
        v16h b1 = load_b_frag(Bs[buf], wn * 32 + 16, lane);
        acc[0][0] = WMMA(a0, b0, acc[0][0]);
        acc[0][1] = WMMA(a0, b1, acc[0][1]);
        acc[1][0] = WMMA(a1, b0, acc[1][0]);
        acc[1][1] = WMMA(a1, b1, acc[1][1]);
        buf ^= 1;
    }

    int hi = lane >> 4, nn = lane & 15;
#pragma unroll
    for (int mi = 0; mi < 2; ++mi) {
#pragma unroll
        for (int ni = 0; ni < 2; ++ni) {
            v8f a = acc[mi][ni];
            int col = colBase + wn * 32 + ni * 16 + nn;
#pragma unroll
            for (int j = 0; j < 8; ++j) {
                int row = rowBase + wm * 32 + mi * 16 + hi * 8 + j;
                int ib = row >> 10, pp2 = row & 1023;
                size_t o = ((size_t)ib * TPAD + 1 + pp2) * DMODEL + col;
                t[o] = a[j] + cb[col] + pe[(size_t)(1 + pp2) * DMODEL + col];
            }
        }
    }
}

// ------------- LayerNorm over dim 256; zeroes padded token rows; f16 out ------------

__global__ __launch_bounds__(256) void ln_kernel(
    const float* __restrict__ in, _Float16* __restrict__ out,
    const float* __restrict__ s, const float* __restrict__ bb,
    int tokens_per_img, int valid_tokens)
{
    __shared__ float red[256];
    int row = blockIdx.x, tid = threadIdx.x;
    size_t base = (size_t)row * DMODEL;
    if ((row % tokens_per_img) >= valid_tokens) { out[base + tid] = (_Float16)0.f; return; }

    float v = in[base + tid];
    red[tid] = v; __syncthreads();
    for (int st = 128; st > 0; st >>= 1) { if (tid < st) red[tid] += red[tid + st]; __syncthreads(); }
    float mean = red[0] * (1.f / 256.f);
    __syncthreads();
    float d = v - mean;
    red[tid] = d * d; __syncthreads();
    for (int st = 128; st > 0; st >>= 1) { if (tid < st) red[tid] += red[tid + st]; __syncthreads(); }
    float r = rsqrtf(red[0] * (1.f / 256.f) + 1e-5f);
    out[base + tid] = (_Float16)(d * r * s[tid] + bb[tid]);
}

// ------------- DFT matrices (exact integer k*n mod N) -------------------------------
// Token DFT is an A-operand -> row-major; dim DFT is a B-operand -> PI layout.

__global__ void init_dft_tok(_Float16* __restrict__ C, _Float16* __restrict__ S) {
    int idx = blockIdx.x * 256 + threadIdx.x;
    if (idx >= TPAD * TPAD) return;
    int k = idx / TPAD, n = idx % TPAD;
    float c = 0.f, s = 0.f;
    if (k < NTOK && n < NTOK) {
        int r = (int)(((long long)k * n) % NTOK);
        float a = 6.28318530717958647692f * (float)r / (float)NTOK;
        c = cosf(a); s = sinf(a);
    }
    C[idx] = (_Float16)c; S[idx] = (_Float16)s;
}

__global__ void init_dft_dim_pi(_Float16* __restrict__ C, _Float16* __restrict__ S) {
    int idx = blockIdx.x * 256 + threadIdx.x;   // 65536, 2N = 512
    int kp = idx >> 9, r = idx & 511;
    int n = r >> 1, k = 2 * kp + (r & 1);
    int rr = (k * n) & 255;
    float a = 6.28318530717958647692f * (float)rr * (1.f / 256.f);
    C[idx] = (_Float16)cosf(a); S[idx] = (_Float16)sinf(a);
}

// ------------- one-time weight conversions into PI f16 ------------------------------

// src row-major K x N (f32) -> dst PI f16
__global__ void cvt_pi_kernel(const float* __restrict__ src, _Float16* __restrict__ dst,
                              int N, long total) {
    long idx = (long)blockIdx.x * 256 + threadIdx.x;
    if (idx >= total) return;
    long twoN = 2L * N;
    long kp = idx / twoN; int r = (int)(idx % twoN);
    int n = r >> 1, k = 2 * (int)kp + (r & 1);
    dst[idx] = (_Float16)src[(size_t)k * N + n];
}

// conv_w is N-major (256 x 768) -> wT PI (K=768, N=256)
__global__ void cvt_convw_pi_kernel(const float* __restrict__ w, _Float16* __restrict__ wT) {
    int idx = blockIdx.x * 256 + threadIdx.x;   // 768*256, 2N = 512
    int kp = idx >> 9, r = idx & 511;
    int n = r >> 1, k = 2 * kp + (r & 1);
    wT[idx] = (_Float16)w[(size_t)n * 768 + k];
}

// head_w (256 x 1000) -> PI f16 padded to N=1024
__global__ void cvt_headw_pi_kernel(const float* __restrict__ w, _Float16* __restrict__ wp) {
    int idx = blockIdx.x * 256 + threadIdx.x;   // 256*1024, 2N = 2048
    int kp = idx >> 11, r = idx & 2047;
    int n = r >> 1, k = 2 * kp + (r & 1);
    wp[idx] = (n < NCLS) ? (_Float16)w[(size_t)k * NCLS + n] : (_Float16)0.f;
}

// ------------- misc small kernels ---------------------------------------------------

__global__ void init_t_edges(const float* __restrict__ cls, const float* __restrict__ pe,
                             float* __restrict__ t) {
    int b = blockIdx.y, u = blockIdx.x, d = threadIdx.x;   // u in [0,64)
    size_t base = (size_t)b * TPAD * DMODEL;
    if (u == 0) t[base + d] = cls[d] + pe[d];
    else        t[base + (size_t)(1024 + u) * DMODEL + d] = 0.f;   // rows 1025..1087
}

__global__ __launch_bounds__(256) void cvt_f32_f16_kernel(const float* __restrict__ s,
                                                          _Float16* __restrict__ d, long n4) {
    long i = (long)blockIdx.x * 256 + threadIdx.x;
    if (i >= n4) return;
    float4 v = ((const float4*)s)[i];
    union { uint2 u; _Float16 h[4]; } o;
    o.h[0] = (_Float16)v.x; o.h[1] = (_Float16)v.y;
    o.h[2] = (_Float16)v.z; o.h[3] = (_Float16)v.w;
    ((uint2*)d)[i] = o.u;
}

__global__ __launch_bounds__(256) void pool_kernel(const float* __restrict__ t,
                                                   float* __restrict__ pooled) {
    int b = blockIdx.x, d = threadIdx.x;
    const float* base = t + (size_t)b * TPAD * DMODEL + d;
    float s = 0.f;
    for (int k = 0; k < NTOK; ++k) s += base[(size_t)k * DMODEL];
    pooled[b * DMODEL + d] = s * (1.0f / (float)NTOK);
}

// softmax over 1000 classes; folds head bias in (logits buffer has ld 1024)
__global__ __launch_bounds__(256) void softmax_kernel(const float* __restrict__ logits,
                                                      const float* __restrict__ hb,
                                                      float* __restrict__ out) {
    __shared__ float red[256];
    int b = blockIdx.x, tid = threadIdx.x;
    const float* L = logits + (size_t)b * 1024;
    float lv[4];
    float m = -3.402823e38f;
#pragma unroll
    for (int j = 0; j < 4; ++j) {
        int i = tid + 256 * j;
        lv[j] = (i < NCLS) ? (L[i] + hb[i]) : -3.402823e38f;
        m = fmaxf(m, lv[j]);
    }
    red[tid] = m; __syncthreads();
    for (int st = 128; st > 0; st >>= 1) { if (tid < st) red[tid] = fmaxf(red[tid], red[tid + st]); __syncthreads(); }
    m = red[0]; __syncthreads();
    float s = 0.f;
#pragma unroll
    for (int j = 0; j < 4; ++j) { int i = tid + 256 * j; if (i < NCLS) s += expf(lv[j] - m); }
    red[tid] = s; __syncthreads();
    for (int st = 128; st > 0; st >>= 1) { if (tid < st) red[tid] += red[tid + st]; __syncthreads(); }
    float inv = 1.f / red[0];
#pragma unroll
    for (int j = 0; j < 4; ++j) {
        int i = tid + 256 * j;
        if (i < NCLS) out[(size_t)b * NCLS + i] = expf(lv[j] - m) * inv;
    }
}

// ------------- host launcher --------------------------------------------------------

extern "C" void kernel_launch(void* const* d_in, const int* in_sizes, int n_in,
                              void* d_out, int out_size, void* d_ws, size_t ws_size,
                              hipStream_t stream) {
    (void)in_sizes; (void)n_in; (void)out_size; (void)ws_size;

    const float* x       = (const float*)d_in[0];
    const float* conv_w  = (const float*)d_in[1];
    const float* conv_b  = (const float*)d_in[2];
    const float* pos_emb = (const float*)d_in[3];
    const float* cls_tok = (const float*)d_in[4];
    const float* ln1_s   = (const float*)d_in[5];
    const float* ln1_b   = (const float*)d_in[6];
    const float* ln2_s   = (const float*)d_in[7];
    const float* ln2_b   = (const float*)d_in[8];
    const float* w1      = (const float*)d_in[9];
    const float* b1      = (const float*)d_in[10];
    const float* w2      = (const float*)d_in[11];
    const float* b2      = (const float*)d_in[12];
    const float* hls     = (const float*)d_in[13];
    const float* hlb     = (const float*)d_in[14];
    const float* head_w  = (const float*)d_in[15];
    const float* head_b  = (const float*)d_in[16];

    // byte-offset workspace allocator
    char* wsb = (char*)d_ws;
    size_t off = 0;
    auto alloc = [&](size_t bytes) -> char* {
        char* r = wsb + off; off = (off + bytes + 255) & ~(size_t)255; return r;
    };

    const size_t TDf  = (size_t)BATCH * TPAD * DMODEL;         // elements
    float*    t    = (float*)alloc(TDf * 4);
    _Float16* h    = (_Float16*)alloc(TDf * 2);
    char*     scr  = alloc((size_t)BATCH * TPAD * 1024 * 2);   // 142.6 MB shared scratch
    _Float16* Ctok = (_Float16*)alloc((size_t)TPAD * TPAD * 2);
    _Float16* Stok = (_Float16*)alloc((size_t)TPAD * TPAD * 2);
    _Float16* Cdim = (_Float16*)alloc(65536 * 2);
    _Float16* Sdim = (_Float16*)alloc(65536 * 2);
    _Float16* wT   = (_Float16*)alloc(768 * 256 * 2);
    _Float16* w1h  = (_Float16*)alloc((size_t)6 * 256 * 1024 * 2);
    _Float16* w2h  = (_Float16*)alloc((size_t)6 * 1024 * 256 * 2);
    _Float16* hwh  = (_Float16*)alloc(256 * 1024 * 2);
    float*    pooled   = (float*)alloc(BATCH * DMODEL * 4);
    _Float16* pooledln = (_Float16*)alloc(BATCH * DMODEL * 2);
    float*    logits   = (float*)alloc(BATCH * 1024 * 4);

    // overlays inside scratch (disjoint in time):
    _Float16* xh   = (_Float16*)scr;                           // patch embed only
    _Float16* tmp1 = (_Float16*)scr;                           // token mixing (PI)
    _Float16* tmp2 = (_Float16*)(scr + TDf * 2);               // (PI)
    float*    mix  = (float*)(scr + TDf * 4);
    _Float16* hid  = (_Float16*)scr;                           // FFN (row-major A)

    auto gemm = [&](const _Float16* A, int lda, long sA,
                    const _Float16* Bm, long sB,
                    void* O, int ldo, long sO,
                    const float* res, long sR, const float* ex, long sE,
                    const float* bias, int M, int N, int K, float alpha, int flags, int Z) {
        dim3 g(N / BN, M / BM, Z);
        gemm_h_kernel<<<g, 256, 0, stream>>>(A, lda, sA, Bm, sB, O, ldo, sO,
                                             res, sR, ex, sE, bias, N, K, alpha, flags);
    };

    // ---- one-time conversions / tables ----
    long nx4 = (long)BATCH * 3 * 512 * 512 / 4;
    cvt_f32_f16_kernel<<<(nx4 + 255) / 256, 256, 0, stream>>>(x, xh, nx4);
    cvt_convw_pi_kernel<<<(768 * 256) / 256, 256, 0, stream>>>(conv_w, wT);
    for (int i = 0; i < 6; ++i) {
        long nw = 256L * 1024;
        cvt_pi_kernel<<<(nw + 255) / 256, 256, 0, stream>>>(
            w1 + (size_t)i * nw, w1h + (size_t)i * nw, 1024, nw);
        cvt_pi_kernel<<<(nw + 255) / 256, 256, 0, stream>>>(
            w2 + (size_t)i * nw, w2h + (size_t)i * nw, 256, nw);
    }
    cvt_headw_pi_kernel<<<(256 * 1024) / 256, 256, 0, stream>>>(head_w, hwh);
    init_dft_tok<<<(TPAD * TPAD + 255) / 256, 256, 0, stream>>>(Ctok, Stok);
    init_dft_dim_pi<<<65536 / 256, 256, 0, stream>>>(Cdim, Sdim);

    // ---- embed ----
    init_t_edges<<<dim3(64, BATCH), 256, 0, stream>>>(cls_tok, pos_emb, t);
    patch_embed_kernel<<<dim3(DMODEL / BN, (BATCH * 1024) / BM), 256, 0, stream>>>(
        xh, wT, conv_b, pos_emb, t);

    const int  Mfull = BATCH * TPAD;          // 69632
    const long td    = (long)TPAD * DMODEL;   // per-image activation stride (elements)

    for (int i = 0; i < 6; ++i) {
        // --- FNet token mixing: t += Ctok@(h@Cdim) - Stok@(h@Sdim) ---
        ln_kernel<<<Mfull, 256, 0, stream>>>(t, h, ln1_s + i * 256, ln1_b + i * 256, TPAD, NTOK);
        gemm(h, 256, 0, Cdim, 0, tmp1, 256, 0,
             nullptr, 0, nullptr, 0, nullptr, Mfull, 256, 256, 1.f, FLAG_OUT_PI, 1);
        gemm(h, 256, 0, Sdim, 0, tmp2, 256, 0,
             nullptr, 0, nullptr, 0, nullptr, Mfull, 256, 256, 1.f, FLAG_OUT_PI, 1);
        gemm(Ctok, TPAD, 0, tmp1, td, mix, 256, td,
             nullptr, 0, nullptr, 0, nullptr, TPAD, 256, TPAD, 1.f, 0, BATCH);
        gemm(Stok, TPAD, 0, tmp2, td, t, 256, td,
             t, td, mix, td, nullptr, TPAD, 256, TPAD, -1.f, 0, BATCH);
        // --- FFN: t += lrelu(h@w1+b1)@w2 + b2 ---
        ln_kernel<<<Mfull, 256, 0, stream>>>(t, h, ln2_s + i * 256, ln2_b + i * 256, TPAD, NTOK);
        gemm(h, 256, 0, w1h + (size_t)i * 256 * 1024, 0, hid, 1024, 0,
             nullptr, 0, nullptr, 0, b1 + i * 1024, Mfull, 1024, 256, 1.f,
             FLAG_LRELU | FLAG_OUTF16, 1);
        gemm(hid, 1024, 0, w2h + (size_t)i * 1024 * 256, 0, t, 256, 0,
             t, 0, nullptr, 0, b2 + i * 256, Mfull, 256, 1024, 1.f, 0, 1);
    }

    // ---- head ----
    pool_kernel<<<BATCH, 256, 0, stream>>>(t, pooled);
    ln_kernel<<<BATCH, 256, 0, stream>>>(pooled, pooledln, hls, hlb, 1, 1);
    gemm(pooledln, 256, 0, hwh, 0, logits, 1024, 0,
         nullptr, 0, nullptr, 0, nullptr, BATCH, 1024, 256, 1.f, 0, 1);
    softmax_kernel<<<BATCH, 256, 0, stream>>>(logits, head_b, (float*)d_out);
}
